// Upsample2DFourierKernel_1125281431750
// MI455X (gfx1250) — matrix operand order repaired and checked by reading, not compiled, not run on hardware
//
#include <hip/hip_runtime.h>

// ---------------------------------------------------------------------------
// FFT-zero-pad 2x upsample == separable circulant GEMMs:
//   out[2m  ,2n  ] = x[m,n]
//   out[2m+1,2n  ] = (A x)[m,n]
//   out[2m  ,2n+1] = (x A^T)[m,n]
//   out[2m+1,2n+1] = (A x A^T)[m,n] - c * (-1)^(m+n)
// A circulant: a_r = (-1)^r cot(pi(2r+1)/(2N))/N ;  c = (1/N^2) sum (-1)^(i+j) x
// GEMMs on v_wmma_f32_16x16x32_f16, hi/lo f16 error-compensated split.
// Circulant WMMA operands come from precomputed per-residue fragment tables
// (one aligned 32B/lane global load per operand); all LDS staging is pure
// vector copies thanks to a one-time transposed copy of x.
// ---------------------------------------------------------------------------

typedef __attribute__((ext_vector_type(16))) _Float16 v16h;
typedef __attribute__((ext_vector_type(8)))  _Float16 v8h;
typedef __attribute__((ext_vector_type(4)))  _Float16 v4h;
typedef __attribute__((ext_vector_type(8)))  float    v8f;

constexpr int kN    = 4096;
constexpr int kMask = 4095;
constexpr int kOutW = 8192;

static __device__ __forceinline__ v8f wmma_f16(v16h a, v16h b, v8f c) {
  return __builtin_amdgcn_wmma_f32_16x16x32_f16(false, a, false, b, (short)0, c,
                                                false, false);
}

// ---- a-vector: a_r = (-1)^r * cot(pi*(2r+1)/(2N)) / N, split into f16 hi/lo -
__global__ void prep_a_kernel(_Float16* __restrict__ avh,
                              _Float16* __restrict__ avl) {
  int r = blockIdx.x * 256 + threadIdx.x;               // 0..4095
  const double PI = 3.14159265358979323846;
  double th   = (double)(2 * r + 1) * (PI / 8192.0);    // pi(2r+1)/(2N)
  double cotv = 1.0 / tan(th);
  float  v    = (float)(((r & 1) ? -cotv : cotv) / 4096.0);
  _Float16 h  = (_Float16)v;
  avh[r] = h;
  avl[r] = (_Float16)(v - (float)h);
}

// ---- per-residue WMMA fragment tables for the circulant operand ------------
// A-layout table (operand A, M=ln):  val[e] = a[(d + ln - K(e)) & mask],
//   K(e) = 8g + (e<8 ? e : e+8)         (ISA 16-bit A 16x32 layout)
// B-layout table (operand B, N=ln):  val[e] = a[(d + ln - 16g - e) & mask]
//   K(e) = 16g + e                      (contiguous k order, same as A in conv_circ_right)
// Table element (d, lane): 16 halves at offset (d*32 + lane)*16.
__global__ void build_frag_tables(const _Float16* __restrict__ avh,
                                  const _Float16* __restrict__ avl,
                                  _Float16* __restrict__ AtH, _Float16* __restrict__ AtL,
                                  _Float16* __restrict__ BtH, _Float16* __restrict__ BtL) {
  int tid  = blockIdx.x * 256 + threadIdx.x;            // 0 .. 4096*32-1
  int lane = tid & 31, d = tid >> 5;
  int g = lane >> 4, ln = lane & 15;
  size_t off = ((size_t)d << 9) + ((size_t)lane << 4);
  v16h ah, al, bh, bl;
#pragma unroll
  for (int e = 0; e < 16; ++e) {
    int K  = (g << 3) + ((e < 8) ? e : (e + 8));
    int ia = (d + ln - K) & kMask;
    ah[e] = avh[ia];
    al[e] = avl[ia];
    int ib = (d + ln - (g << 4) - e) & kMask;
    bh[e] = avh[ib];
    bl[e] = avl[ib];
  }
  *(v16h*)&AtH[off] = ah;
  *(v16h*)&AtL[off] = al;
  *(v16h*)&BtH[off] = bh;
  *(v16h*)&BtL[off] = bl;
}

// ---- deterministic checkerboard reduction: c = (1/N^2) sum (-1)^(i+j) x ----
__global__ void reduce_cb1(const float* __restrict__ x, float* __restrict__ partials) {
  int t = threadIdx.x, b = blockIdx.x;
  float s = 0.0f;
  for (int it = 0; it < 64; ++it) {
    int idx  = (b * 256 + t) + it * (1024 * 256);       // covers 2^24 elements
    float v  = x[idx];
    int  par = ((idx >> 12) ^ idx) & 1;                 // parity of i+j
    s += par ? -v : v;
  }
  __shared__ float sm[256];
  sm[t] = s; __syncthreads();
  for (int off = 128; off > 0; off >>= 1) {
    if (t < off) sm[t] += sm[t + off];
    __syncthreads();
  }
  if (t == 0) partials[b] = sm[0];
}

__global__ void reduce_cb2(const float* __restrict__ partials, float* __restrict__ c) {
  int t = threadIdx.x;
  float s = 0.0f;
  for (int i = t; i < 1024; i += 256) s += partials[i];
  __shared__ float sm[256];
  sm[t] = s; __syncthreads();
  for (int off = 128; off > 0; off >>= 1) {
    if (t < off) sm[t] += sm[t + off];
    __syncthreads();
  }
  if (t == 0) *c = sm[0] * (1.0f / 16777216.0f);        // / N^2
}

// ---- split x into f16 hi/lo (row-major AND transposed) + (even,even) copy --
__global__ void transpose_split(const float* __restrict__ x,
                                _Float16* __restrict__ xh,  _Float16* __restrict__ xl,
                                _Float16* __restrict__ xth, _Float16* __restrict__ xtl,
                                float* __restrict__ out) {
  __shared__ float S[64 * 65];
  const int t  = threadIdx.x;
  const int i0 = (blockIdx.x & 63) * 64;
  const int j0 = (blockIdx.x >> 6) * 64;
#pragma unroll
  for (int it = 0; it < 16; ++it) {
    int lin = t + 256 * it;                             // 0..4095
    int r = lin >> 6, c = lin & 63;
    size_t gi = (size_t)(i0 + r) * kN + j0 + c;
    float v = x[gi];
    _Float16 h = (_Float16)v;
    xh[gi] = h;
    xl[gi] = (_Float16)(v - (float)h);
    out[(size_t)(2 * (i0 + r)) * kOutW + 2 * (j0 + c)] = v;   // y[2i,2j]=x[i,j]
    S[r * 65 + c] = v;
  }
  __syncthreads();
#pragma unroll
  for (int it = 0; it < 4; ++it) {
    int lin = t + 256 * it;                             // 0..1023
    int rr = lin >> 4;                                  // j within tile
    int c4 = (lin & 15) * 4;                            // i start
    v4h hv, lv;
#pragma unroll
    for (int q = 0; q < 4; ++q) {
      float v = S[(c4 + q) * 65 + rr];
      _Float16 h = (_Float16)v;
      hv[q] = h;
      lv[q] = (_Float16)(v - (float)h);
    }
    *(v4h*)&xth[(size_t)(j0 + rr) * kN + i0 + c4] = hv;
    *(v4h*)&xtl[(size_t)(j0 + rr) * kN + i0 + c4] = lv;
  }
}

// ---- O = circ(a) * X : O[i,j] = sum_k a[(i-k)&M] X[k,j] --------------------
// Reads X^T (so the k reduction axis is memory-contiguous).
// Block 256 thr (8 waves): tile 128(M) x 128(N); wave = 16 rows x 128 cols.
// Circulant A operand: one v16h table load per wave/k-step (reused 8x).
__global__ void __launch_bounds__(256)
conv_circ_left(const _Float16* __restrict__ AtH, const _Float16* __restrict__ AtL,
               const _Float16* __restrict__ XTh, const _Float16* __restrict__ XTl,
               float* __restrict__ outq,
               _Float16* __restrict__ Ph, _Float16* __restrict__ Pl) {
  __shared__ alignas(16) _Float16 Bh_s[128 * 40];       // [n][k], pitch 40 halves
  __shared__ alignas(16) _Float16 Bl_s[128 * 40];

  const int t = threadIdx.x;
  const int w = t >> 5, lane = t & 31;
  const int g = lane >> 4, ln = lane & 15;
  const int i0 = blockIdx.y * 128, j0 = blockIdx.x * 128;
  const int m0 = i0 + 16 * w;
  const int ns = t >> 1;                                // staging row 0..127
  const int kh = (t & 1) * 16;                          // staging k-half

  v8f acc[8] = {};

  for (int k0 = 0; k0 < kN; k0 += 32) {
    __syncthreads();
    {  // stage X^T[j0+n, k0:k0+32] -> [n][k] (pure vector copies)
      const v8h* gh = (const v8h*)&XTh[(size_t)(j0 + ns) * kN + k0 + kh];
      const v8h* gl = (const v8h*)&XTl[(size_t)(j0 + ns) * kN + k0 + kh];
      *(v8h*)&Bh_s[ns * 40 + kh]     = gh[0];
      *(v8h*)&Bh_s[ns * 40 + kh + 8] = gh[1];
      *(v8h*)&Bl_s[ns * 40 + kh]     = gl[0];
      *(v8h*)&Bl_s[ns * 40 + kh + 8] = gl[1];
      if (k0 + 32 < kN)
        __builtin_prefetch(&XTh[(size_t)(j0 + ns) * kN + k0 + 32 + kh], 0, 0);
    }
    __syncthreads();

    // circulant A fragments straight from the residue table
    const int dA = (m0 - k0) & kMask;
    const size_t offA = ((size_t)dA << 9) + ((size_t)lane << 4);
    v16h Ah = *(const v16h*)&AtH[offA];
    v16h Al = *(const v16h*)&AtL[offA];

#pragma unroll
    for (int nt = 0; nt < 8; ++nt) {
      const _Float16* bph = &Bh_s[(nt * 16 + ln) * 40 + (g << 3)];
      const _Float16* bpl = &Bl_s[(nt * 16 + ln) * 40 + (g << 3)];
      v8h bh0 = *(const v8h*)bph, bh1 = *(const v8h*)(bph + 16);
      v8h bl0 = *(const v8h*)bpl, bl1 = *(const v8h*)(bpl + 16);
      v16h Bh, Bl;
#pragma unroll
      for (int e = 0; e < 8; ++e) {
        Bh[e] = bh0[e]; Bh[e + 8] = bh1[e];
        Bl[e] = bl0[e]; Bl[e + 8] = bl1[e];
      }
      acc[nt] = wmma_f16(Ah, Bh, acc[nt]);              // hi*hi
      acc[nt] = wmma_f16(Ah, Bl, acc[nt]);              // hi*lo
      acc[nt] = wmma_f16(Al, Bh, acc[nt]);              // lo*hi
    }
  }

  // scatter into (odd,even) quadrant + write f16 split of P (row-major)
#pragma unroll
  for (int nt = 0; nt < 8; ++nt) {
#pragma unroll
    for (int v = 0; v < 8; ++v) {
      int row = m0 + v + 8 * g;                         // ISA C/D layout
      int col = j0 + nt * 16 + ln;
      float val = acc[nt][v];
      outq[(size_t)row * (2 * kOutW) + (size_t)col * 2] = val;
      _Float16 h = (_Float16)val;
      Ph[(size_t)row * kN + col] = h;
      Pl[(size_t)row * kN + col] = (_Float16)(val - (float)h);
    }
  }
}

// ---- O = X * circ(a)^T : O[i,j] = sum_k X[i,k] a[(j-k)&M] ------------------
// X row-major (k contiguous). Circulant is the B operand, from residue table.
__global__ void __launch_bounds__(256)
conv_circ_right(const _Float16* __restrict__ BtH, const _Float16* __restrict__ BtL,
                const _Float16* __restrict__ Xh,  const _Float16* __restrict__ Xl,
                float* __restrict__ outq, const float* __restrict__ cptr) {
  __shared__ alignas(16) _Float16 Th_s[128 * 40];       // [m][k], pitch 40 halves
  __shared__ alignas(16) _Float16 Tl_s[128 * 40];

  const int t = threadIdx.x;
  const int w = t >> 5, lane = t & 31;
  const int g = lane >> 4, ln = lane & 15;
  const int i0 = blockIdx.y * 128, j0 = blockIdx.x * 128;
  const int ms = t >> 1;                                // staging row 0..127
  const int kh = (t & 1) * 16;                          // staging k-half

  v8f acc[8] = {};

  for (int k0 = 0; k0 < kN; k0 += 32) {
    __syncthreads();
    {  // stage X[i0+m, k0:k0+32] -> [m][k] (pure vector copies)
      const v8h* gh = (const v8h*)&Xh[(size_t)(i0 + ms) * kN + k0 + kh];
      const v8h* gl = (const v8h*)&Xl[(size_t)(i0 + ms) * kN + k0 + kh];
      *(v8h*)&Th_s[ms * 40 + kh]     = gh[0];
      *(v8h*)&Th_s[ms * 40 + kh + 8] = gh[1];
      *(v8h*)&Tl_s[ms * 40 + kh]     = gl[0];
      *(v8h*)&Tl_s[ms * 40 + kh + 8] = gl[1];
      if (k0 + 32 < kN)
        __builtin_prefetch(&Xh[(size_t)(i0 + ms) * kN + k0 + 32 + kh], 0, 0);
    }
    __syncthreads();

    // dense A fragment from staged rows (contiguous k order, k = 16g + e, as the B table)
    const _Float16* aph = &Th_s[(16 * w + ln) * 40];
    const _Float16* apl = &Tl_s[(16 * w + ln) * 40];
    v8h ah0 = *(const v8h*)(aph + 16 * g), ah1 = *(const v8h*)(aph + 16 * g + 8);
    v8h al0 = *(const v8h*)(apl + 16 * g), al1 = *(const v8h*)(apl + 16 * g + 8);
    v16h Ah, Al;
#pragma unroll
    for (int e = 0; e < 8; ++e) {
      Ah[e] = ah0[e]; Ah[e + 8] = ah1[e];
      Al[e] = al0[e]; Al[e + 8] = al1[e];
    }

#pragma unroll
    for (int nt = 0; nt < 8; ++nt) {
      // circulant B fragment straight from the residue table
      const int dB = (j0 + nt * 16 - k0) & kMask;
      const size_t offB = ((size_t)dB << 9) + ((size_t)lane << 4);
      v16h Bh = *(const v16h*)&BtH[offB];
      v16h Bl = *(const v16h*)&BtL[offB];
      acc[nt] = wmma_f16(Ah, Bh, acc[nt]);
      acc[nt] = wmma_f16(Ah, Bl, acc[nt]);
      acc[nt] = wmma_f16(Al, Bh, acc[nt]);
    }
  }

  const float cs = cptr ? *cptr : 0.0f;                 // checkerboard correction
  const int m0 = i0 + 16 * w;
#pragma unroll
  for (int nt = 0; nt < 8; ++nt) {
#pragma unroll
    for (int v = 0; v < 8; ++v) {
      int row = m0 + v + 8 * g;
      int col = j0 + nt * 16 + ln;
      float val = acc[nt][v] - cs * (((row + col) & 1) ? -1.0f : 1.0f);
      outq[(size_t)row * (2 * kOutW) + (size_t)col * 2] = val;
    }
  }
}

// ---------------------------------------------------------------------------
extern "C" void kernel_launch(void* const* d_in, const int* in_sizes, int n_in,
                              void* d_out, int out_size, void* d_ws, size_t ws_size,
                              hipStream_t stream) {
  (void)in_sizes; (void)n_in; (void)out_size; (void)ws_size;
  const float* x   = (const float*)d_in[0];   // 4096x4096 f32 (factor token: f=2)
  float*       out = (float*)d_out;           // 8192x8192 f32
  char*        ws  = (char*)d_ws;

  constexpr size_t kMat  = (size_t)kN * kN;             // 16M halves = 32 MB
  constexpr size_t kTab  = (size_t)kN * 512;            // 2M halves  =  4 MB

  _Float16* avh      = (_Float16*)(ws);                    //  8 KB
  _Float16* avl      = (_Float16*)(ws + (8u  << 10));      //  8 KB
  float*    partials = (float*)   (ws + (16u << 10));      //  4 KB
  float*    cval     = (float*)   (ws + (24u << 10));      //  4 B
  _Float16* AtH      = (_Float16*)(ws + (64u << 10));      //  4 MB
  _Float16* AtL      = AtH + kTab;                         //  4 MB
  _Float16* BtH      = AtL + kTab;                         //  4 MB
  _Float16* BtL      = BtH + kTab;                         //  4 MB
  _Float16* xh       = BtL + kTab;                         // 32 MB
  _Float16* xl       = xh  + kMat;                         // 32 MB
  _Float16* xth      = xl  + kMat;                         // 32 MB
  _Float16* xtl      = xth + kMat;                         // 32 MB
  _Float16* ph       = xtl + kMat;                         // 32 MB
  _Float16* pl       = ph  + kMat;                         // 32 MB

  prep_a_kernel<<<16, 256, 0, stream>>>(avh, avl);
  build_frag_tables<<<512, 256, 0, stream>>>(avh, avl, AtH, AtL, BtH, BtL);
  reduce_cb1<<<1024, 256, 0, stream>>>(x, partials);
  reduce_cb2<<<1, 256, 0, stream>>>(partials, cval);
  transpose_split<<<4096, 256, 0, stream>>>(x, xh, xl, xth, xtl, out);

  // P = A x          -> (odd,even) quadrant + f16 split kept for R
  conv_circ_left<<<dim3(32, 32), 256, 0, stream>>>(AtH, AtL, xth, xtl,
                                                   out + kOutW, ph, pl);
  // Q = x A^T        -> (even,odd) quadrant
  conv_circ_right<<<dim3(32, 32), 256, 0, stream>>>(BtH, BtL, xh, xl,
                                                    out + 1, nullptr);
  // R = P A^T - c*cb -> (odd,odd) quadrant
  conv_circ_right<<<dim3(32, 32), 256, 0, stream>>>(BtH, BtL, ph, pl,
                                                    out + kOutW + 1, cval);
}
